// MultiHeadedSelfAttention_40029095198755
// MI455X (gfx1250) — matrix operand order repaired and checked by reading, not compiled, not run on hardware
//
#include <hip/hip_runtime.h>

// ---------------------------------------------------------------------------
// MultiHeadedSelfAttention for MI455X (gfx1250, wave32, WMMA, TDM)
// B=4, S=2048, Dmodel=1024, H=16, head_dim=64
//
// Stage 1: q16 = f16(pre_q @ Wq^T + bq)        row-major  [8192,1024]
//          k16 = f16(pre_vk @ Wk^T + bk)       row-major  [8192,1024]
//          vt16 = f16(pre_vk @ Wv^T + bv)      transposed [b][h][d=64][s=2048]
//          W tiles staged to LDS via Tensor Data Mover (double-buffered DMA).
// Stage 2: per (b,h,qtile16): scores via WMMA, masked softmax + pooled sigmoid
//          gate through 205 KB LDS (CDNA5-only capacity), PV via WMMA.
// All per-lane addressing is uniform-base + 32-bit offset so loads lower to
// the saddr+voffset global form (no per-lane u64 VALU chains).
// ---------------------------------------------------------------------------

typedef _Float16 v16h __attribute__((ext_vector_type(16)));
typedef _Float16 v8h  __attribute__((ext_vector_type(8)));
typedef _Float16 v4h  __attribute__((ext_vector_type(4)));
typedef float    v8f  __attribute__((ext_vector_type(8)));
typedef float    v4f  __attribute__((ext_vector_type(4)));
typedef unsigned int u32x4 __attribute__((ext_vector_type(4)));
typedef int          i32x8 __attribute__((ext_vector_type(8)));
typedef int          i32x4 __attribute__((ext_vector_type(4)));

union V16H  { v16h v; v8h h[2]; };
union V16HQ { v16h v; v4h q[4]; };

constexpr int BATCH  = 4;
constexpr int SEQ    = 2048;
constexpr int DMODEL = 1024;
constexpr int NH     = 16;
constexpr int HD     = 64;
constexpr int MROWS  = BATCH * SEQ;      // 8192
constexpr int WROW   = 36;               // padded LDS row pitch (32 data + 4 pad dwords)

#define WMMA_F16(Afrag, Bfrag, Cfrag) \
  __builtin_amdgcn_wmma_f32_16x16x32_f16(false, (Afrag), false, (Bfrag), (short)0, (Cfrag), false, false)

#if defined(__HIP_DEVICE_COMPILE__) && __has_builtin(__builtin_amdgcn_tensor_load_to_lds)
#define HAVE_TDM 1
#else
#define HAVE_TDM 0
#endif

// ---------------------------------------------------------------------------
// TDM: DMA a 2D f32 tile (tileDim0 contiguous dwords x tileDim1 rows, row
// stride strideElts) from global into LDS at ldsAddr, padding +4 dwords per
// 32 stored dwords (row pitch 36 dwords).  D# per CDNA5 ISA ch.8.
// ---------------------------------------------------------------------------
#if HAVE_TDM
__device__ __forceinline__ void tdm_load_2d_f32(unsigned ldsAddr, const float* gptr,
                                                unsigned tileDim0, unsigned tileDim1,
                                                unsigned strideElts) {
  const unsigned long long ga = (unsigned long long)(const void*)gptr;
  u32x4 g0;
  g0[0] = 1u;                                             // count=1, user mode
  g0[1] = ldsAddr;                                        // lds_addr (bytes)
  g0[2] = (unsigned)(ga & 0xFFFFFFFFu);                   // global_addr[31:0]
  g0[3] = (unsigned)((ga >> 32) & 0x01FFFFFFu) | (2u << 30); // addr[56:32] | type=2

  const unsigned td0 = 1u << 20, td1 = 1u << 20;          // huge tensor dims: no OOB
  i32x8 g1;
  g1[0] = (int)((2u << 16)        // data_size = 4B
              | (1u << 20)        // pad_enable
              | (4u << 22)        // pad_interval: code 4 = every 32 dwords
              | (3u << 25));      // pad_amount:   code 3 = 4 dwords
  g1[1] = (int)((td0 & 0xFFFFu) << 16);                   // tensor_dim0[15:0]
  g1[2] = (int)((td0 >> 16) | ((td1 & 0xFFFFu) << 16));   // td0[31:16] | td1[15:0]
  g1[3] = (int)((td1 >> 16) | (tileDim0 << 16));          // td1[31:16] | tile_dim0
  g1[4] = (int)(tileDim1);                                // tile_dim1 | tile_dim2=0
  g1[5] = (int)strideElts;                                // tensor_dim0_stride[31:0]
  g1[6] = 0;                                              // stride0[47:32] | stride1[15:0]
  g1[7] = 0;
  i32x4 g2 = {0, 0, 0, 0};
  i32x4 g3 = {0, 0, 0, 0};
  i32x8 g4 = {0, 0, 0, 0, 0, 0, 0, 0};
  __builtin_amdgcn_tensor_load_to_lds(g0, g1, g2, g3, g4, 0);
}
#endif

// Stage W[n0..n0+127][kb..kb+31] (f32) into LDS with row pitch WROW dwords.
__device__ __forceinline__ void stage_w(const float* __restrict__ W, int n0, int kb,
                                        float* dst, int tid, int wave) {
#if HAVE_TDM
  if (wave == 0) {
    const unsigned ldsAddr = (unsigned)(unsigned long long)(const void*)dst;
    tdm_load_2d_f32(ldsAddr, W + (size_t)n0 * DMODEL + kb, 32u, 128u, (unsigned)DMODEL);
  }
#else
  for (int e = tid; e < 128 * 32; e += 256) {
    const int r = e >> 5, c = e & 31;
    dst[r * WROW + c] = W[(size_t)(n0 + r) * DMODEL + kb + c];
  }
#endif
}

__device__ __forceinline__ void wait_tdm() {
#if HAVE_TDM
  __builtin_amdgcn_s_wait_tensorcnt(0);
#endif
}

// ---------------------------------------------------------------------------
// Projection GEMM: C[m,n] = sum_k A[m,k] * W[n,k] + bias[n]   (f32 in, f16 out)
// Block = 256 threads (8 waves), 128m x 128n tile, K stepped by 32.
// W tiles double-buffered in LDS via TDM; A-fragments straight from global.
// ---------------------------------------------------------------------------
template <bool TRANSPOSED_OUT>
__global__ __launch_bounds__(256) void gemm_proj(const float* __restrict__ A,
                                                 const float* __restrict__ W,
                                                 const float* __restrict__ bias,
                                                 _Float16* __restrict__ out) {
  __shared__ float wtile[2][128 * WROW];

  const int tid  = threadIdx.x;
  const int wave = tid >> 5;
  const int lane = tid & 31;
  const int half16 = lane >> 4;           // 0 or 1
  const int l15    = lane & 15;

  const int m0 = blockIdx.y * 128 + wave * 16;
  const int n0 = blockIdx.x * 128;
  const int ko = half16 * 16;             // A-frag K sub-offset

  // Uniform base + 32-bit per-lane offset (lowered to saddr+voffset form).
  const float* abase = A + (size_t)m0 * DMODEL;
  const int aoff = l15 * DMODEL;          // lane's A row offset

  v8f acc[8] = {};

  stage_w(W, n0, 0, wtile[0], tid, wave);
  if (wave == 0) wait_tdm();
  __syncthreads();

  for (int kb = 0; kb < DMODEL; kb += 32) {
    const int buf = (kb >> 5) & 1;
    const bool more = (kb + 32) < DMODEL;
    if (more) stage_w(W, n0, kb + 32, wtile[buf ^ 1], tid, wave);

    const float* ap = abase + (aoff + kb);
    if (more) __builtin_prefetch(ap + 32, 0, 0);          // global_prefetch_b8

    V16H a;
    a.h[0] = __builtin_convertvector(*(const v8f*)(ap + ko), v8h);
    a.h[1] = __builtin_convertvector(*(const v8f*)(ap + ko + 8), v8h);

#pragma unroll
    for (int t = 0; t < 8; ++t) {
      // B[k,n] = W[n,k]: lane holds column n = n0+t*16+l15, 16 contiguous k.
      const float* wr = &wtile[buf][(t * 16 + l15) * WROW + half16 * 16];
      V16HQ bb;
#pragma unroll
      for (int p = 0; p < 4; ++p)
        bb.q[p] = __builtin_convertvector(*(const v4f*)(wr + p * 4), v4h);
      acc[t] = WMMA_F16(a.v, bb.v, acc[t]);
    }

    if (wave == 0 && more) wait_tdm();
    __syncthreads();
  }

#pragma unroll
  for (int t = 0; t < 8; ++t) {
    const int n  = n0 + t * 16 + l15;
    const float bn = bias[n];
#pragma unroll
    for (int i = 0; i < 8; ++i) {
      const int m = m0 + half16 * 8 + i;  // D layout: lanes<16 rows 0..7, lanes>=16 rows 8..15
      const float val = acc[t][i] + bn;
      if (!TRANSPOSED_OUT) {
        out[m * DMODEL + n] = (_Float16)val;              // 32-bit offset (max 2^23)
      } else {
        const int b = m >> 11, s = m & (SEQ - 1);
        const int h = n >> 6,  d = n & (HD - 1);
        out[(((b * NH + h) * HD + d) << 11) + s] = (_Float16)val;  // vT[b][h][d][s]
      }
    }
  }
}

// ---------------------------------------------------------------------------
// Attention: one block per (qtile16, h, b). 8 waves, ~205KB dynamic LDS.
// ---------------------------------------------------------------------------
constexpr int SMEM_SCORES = 16 * SEQ * 4;          // 131072  f32 masked scaled scores
constexpr int SMEM_P      = 16 * SEQ * 2;          //  65536  f16 exp(P)
constexpr int SMEM_PART   = 8 * 8 * 32 * 4;        //   8192  split-K partials
constexpr int SMEM_POOL   = 8 * 16 * 4;            //    512
constexpr int SMEM_MAX    = 8 * 16 * 4;            //    512
constexpr int SMEM_SCALE  = 16 * 4;                //     64
constexpr int SMEM_BYTES  = SMEM_SCORES + SMEM_P + SMEM_PART + SMEM_POOL + SMEM_MAX + SMEM_SCALE;

__global__ __launch_bounds__(256) void attn_kernel(const _Float16* __restrict__ q16,
                                                   const _Float16* __restrict__ k16,
                                                   const _Float16* __restrict__ vt16,
                                                   const float* __restrict__ masks,
                                                   const float* __restrict__ counts,
                                                   const float* __restrict__ gain,
                                                   const float* __restrict__ biasg,
                                                   float* __restrict__ out) {
  extern __shared__ char smem[];
  float*    s_scores = (float*)smem;                                   // [16][2048]
  _Float16* s_p      = (_Float16*)(smem + SMEM_SCORES);                // [16][2048]
  float*    s_part   = (float*)(smem + SMEM_SCORES + SMEM_P);          // [8][8][32]
  float*    s_pool   = s_part + 8 * 8 * 32;                            // [8][16]
  float*    s_maxa   = s_pool + 8 * 16;                                // [8][16]
  float*    s_scale  = s_maxa + 8 * 16;                                // [16]

  const int qt = blockIdx.x, h = blockIdx.y, b = blockIdx.z;
  const int tid = threadIdx.x, wave = tid >> 5, lane = tid & 31;
  const int half16 = lane >> 4, l15 = lane & 15;

  const float cnt = counts[b];
  const float rs  = rsqrtf(cnt);
  const float* maskb = masks + b * SEQ;

  // Wave-uniform base pointers; all per-lane offsets are 32-bit.
  const _Float16* qbase = q16 + (size_t)(b * SEQ + qt * 16) * DMODEL + h * HD;
  const _Float16* kbase = k16 + (size_t)(b * SEQ) * DMODEL + h * HD;
  const _Float16* vbase = vt16 + (size_t)((b * NH + h) * HD) * SEQ;
  float* obase = out + (size_t)(b * SEQ + qt * 16) * DMODEL + h * HD;

  // ---- Q A-fragments (16 queries x 64 dims = 2 x K32 frags), f16 row-major --
  const int ko = half16 * 16;
  const _Float16* qrow = qbase + l15 * DMODEL;
  V16H a0, a1;
  a0.h[0] = *(const v8h*)(qrow + ko);       a0.h[1] = *(const v8h*)(qrow + 8 + ko);
  a1.h[0] = *(const v8h*)(qrow + 32 + ko);  a1.h[1] = *(const v8h*)(qrow + 40 + ko);

  float pacc[8], pmax[8];
#pragma unroll
  for (int i = 0; i < 8; ++i) { pacc[i] = 0.0f; pmax[i] = -1e30f; }

  // ---- Pass 1: scores = q.k^T; mask/scale into LDS; pooled + max in regs ----
  for (int it = 0; it < 16; ++it) {
    const int key0 = (wave + it * 8) * 16;
    const int koff = (key0 + l15) * DMODEL + half16 * 16;   // 32-bit lane offset
    v16h b0 = *(const v16h*)(kbase + koff);                 // B[kdim,key]: contiguous dim
    v16h b1 = *(const v16h*)(kbase + koff + 32);
    v8f c = {};
    c = WMMA_F16(a0.v, b0, c);
    c = WMMA_F16(a1.v, b1, c);

    const int key = key0 + l15;           // D layout: this lane's key column
    const float mk = maskb[key];
#pragma unroll
    for (int i = 0; i < 8; ++i) {
      const float sraw = c[i];
      pacc[i] += sraw * mk;
      const float sv = (mk == 0.0f) ? -1e30f : sraw * rs;
      pmax[i] = fmaxf(pmax[i], sv);
      const int m = half16 * 8 + i;
      s_scores[m * SEQ + key] = sv;
    }
  }
  // reduce pooled/max across the 16-lane half (each half holds rows 0-7 / 8-15)
#pragma unroll
  for (int i = 0; i < 8; ++i) {
    float p = pacc[i], mx = pmax[i];
    for (int off = 1; off < 16; off <<= 1) {
      p  += __shfl_xor(p, off);
      mx  = fmaxf(mx, __shfl_xor(mx, off));
    }
    if (l15 == 0) {
      const int m = half16 * 8 + i;
      s_pool[wave * 16 + m] = p;
      s_maxa[wave * 16 + m] = mx;
    }
  }
  __syncthreads();

  // ---- Pass 2: per-row stats; exp into f16 P; fold sigmoid gate into scale --
  {
    const int m = wave * 2 + half16;      // 8 waves x 2 rows = 16 rows
    float mx = -1e30f, pool = 0.0f;
    for (int w = 0; w < 8; ++w) {
      mx    = fmaxf(mx, s_maxa[w * 16 + m]);
      pool += s_pool[w * 16 + m];
    }
    float z = 0.0f;
    const float* srow = s_scores + m * SEQ;
    _Float16*    prow = s_p      + m * SEQ;
    for (int j = 0; j < SEQ / 16; ++j) {
      const int kk = l15 + j * 16;
      const float e = __expf(srow[kk] - mx);
      z += e;
      prow[kk] = (_Float16)e;
    }
    for (int off = 1; off < 16; off <<= 1) z += __shfl_xor(z, off);
    const float pooled = pool / cnt;
    const float sig = 1.0f / (1.0f + __expf(-(pooled * gain[h] + biasg[h])));
    if (l15 == 0) s_scale[m] = sig / z;
  }
  __syncthreads();

  // ---- Pass 3: out = (P @ V) * scale.  4 d-tiles x 2-way split-K ----------
  {
    const int dt = wave & 3, kh = wave >> 2;
    const int voff0 = (dt * 16 + l15) * SEQ + half16 * 16;  // 32-bit lane offset
    v8f acc = {};
    for (int ch = 0; ch < 32; ++ch) {
      const int kb2 = kh * 1024 + ch * 32;
      const _Float16* prow = s_p + l15 * SEQ + kb2;
      V16H a;
      a.h[0] = *(const v8h*)(prow + ko);
      a.h[1] = *(const v8h*)(prow + 8 + ko);
      v16h bv = *(const v16h*)(vbase + voff0 + kb2);        // B[key,d]: contiguous keys (vT)
      acc = WMMA_F16(a.v, bv, acc);
    }
#pragma unroll
    for (int i = 0; i < 8; ++i) s_part[(wave * 8 + i) * 32 + lane] = acc[i];
  }
  __syncthreads();

  if (wave < 4) {
#pragma unroll
    for (int i = 0; i < 8; ++i) {
      const float val = s_part[(wave * 8 + i) * 32 + lane] +
                        s_part[((wave + 4) * 8 + i) * 32 + lane];
      const int m = half16 * 8 + i;
      const int d = wave * 16 + l15;
      obase[m * DMODEL + d] = val * s_scale[m];
    }
  }
}

// ---------------------------------------------------------------------------
extern "C" void kernel_launch(void* const* d_in, const int* in_sizes, int n_in,
                              void* d_out, int out_size, void* d_ws, size_t ws_size,
                              hipStream_t stream) {
  const float* pre_vk  = (const float*)d_in[0];
  const float* pre_q   = (const float*)d_in[1];
  const float* vmasks  = (const float*)d_in[2];
  const float* vcounts = (const float*)d_in[3];
  const float* Wq = (const float*)d_in[4];
  const float* bq = (const float*)d_in[5];
  const float* Wk = (const float*)d_in[6];
  const float* bk = (const float*)d_in[7];
  const float* Wv = (const float*)d_in[8];
  const float* bv = (const float*)d_in[9];
  const float* gain  = (const float*)d_in[10];
  const float* biasg = (const float*)d_in[11];

  _Float16* q16  = (_Float16*)d_ws;                       // 16 MB
  _Float16* k16  = q16 + (size_t)MROWS * DMODEL;          // 16 MB
  _Float16* vt16 = k16 + (size_t)MROWS * DMODEL;          // 16 MB

  dim3 ggrid(DMODEL / 128, MROWS / 128);                  // (8, 64)
  gemm_proj<false><<<ggrid, 256, 0, stream>>>(pre_q,  Wq, bq, q16);
  gemm_proj<false><<<ggrid, 256, 0, stream>>>(pre_vk, Wk, bk, k16);
  gemm_proj<true ><<<ggrid, 256, 0, stream>>>(pre_vk, Wv, bv, vt16);

  (void)hipFuncSetAttribute((const void*)attn_kernel,
                            hipFuncAttributeMaxDynamicSharedMemorySize, SMEM_BYTES);
  attn_kernel<<<dim3(SEQ / 16, NH, BATCH), 256, SMEM_BYTES, stream>>>(
      q16, k16, vt16, vmasks, vcounts, gain, biasg, (float*)d_out);
}
